// VectorQuantizer_79276506349613
// MI455X (gfx1250) — matrix operand order repaired and checked by reading, not compiled, not run on hardware
//
#include <hip/hip_runtime.h>
#include <hip/hip_bf16.h>

#define NUM_EMB 2048
#define DIM     128
#define NVEC    32768     // 32 * 32 * 32
#define OUTQ    4194304   // 32 * 128 * 32 * 32
#define LOSS_OFF OUTQ
#define IDXF_OFF (OUTQ + 1)

// LDS staging geometry for the main kernel: 64 codes per stage, padded rows.
#define STAGE_CODES 64
#define NSTAGES     (NUM_EMB / STAGE_CODES)   // 32
#define ROW_DW      68                        // 64 data dwords + 4 pad (16B-aligned rows)

typedef __bf16 bf16_t;
typedef bf16_t bf16x16 __attribute__((ext_vector_type(16)));
typedef float  f32x8   __attribute__((ext_vector_type(8)));

union FragA { bf16x16 v; unsigned int u[8]; };
union FragB { bf16x16 v; uint4 q[2]; };

#if defined(__has_builtin)
#if __has_builtin(__builtin_amdgcn_tensor_load_to_lds)
#define USE_TDM 1
#endif
#if __has_builtin(__builtin_amdgcn_global_load_async_to_lds_b128)
#define USE_ASYNC_LDS 1
#endif
#endif

#if defined(__has_builtin) && __has_builtin(__builtin_amdgcn_s_wait_asynccnt)
#define WAIT_ASYNC() __builtin_amdgcn_s_wait_asynccnt(0)
#else
#define WAIT_ASYNC() asm volatile("s_wait_asynccnt 0x0" ::: "memory")
#endif

// Async-builtin pointer types (from hipcc diagnostic): int4 in AS1 / AS3.
typedef int v4i_vs __attribute__((vector_size(16)));
typedef __attribute__((address_space(1))) v4i_vs global_v4i;
typedef __attribute__((address_space(3))) v4i_vs lds_v4i;
typedef __attribute__((address_space(3))) void lds_void;

// TDM descriptor groups (this toolchain: 6-arg builtin
//   (u32x4 g0, i32x8 g1, i32x4 g2, i32x4 g3, i32x8, i32 cpol)).
typedef unsigned int u32x4 __attribute__((ext_vector_type(4)));
typedef int          i32x8 __attribute__((ext_vector_type(8)));
typedef int          i32x4 __attribute__((ext_vector_type(4)));

__device__ __forceinline__ unsigned short f2bf(float f) {
  unsigned int u = __float_as_uint(f);
  u += 0x7FFFu + ((u >> 16) & 1u);   // round-to-nearest-even
  return (unsigned short)(u >> 16);
}

// ---------------------------------------------------------------------------
// Kernel 1: codebook -> bf16, plus per-code squared norms
// ---------------------------------------------------------------------------
__global__ void __launch_bounds__(128) prep_emb(const float* __restrict__ emb,
                                                unsigned short* __restrict__ e16,
                                                float* __restrict__ enorm) {
  const int k = blockIdx.x;
  const int d = threadIdx.x;
  const float v = emb[k * DIM + d];
  e16[k * DIM + d] = f2bf(v);
  float s = v * v;
  #pragma unroll
  for (int m = 16; m >= 1; m >>= 1) s += __shfl_xor(s, m);
  __shared__ float red[4];
  if ((d & 31) == 0) red[d >> 5] = s;
  __syncthreads();
  if (d == 0) enorm[k] = red[0] + red[1] + red[2] + red[3];
}

// ---------------------------------------------------------------------------
// Kernel 2: (B, D, H, W) f32 -> row-major (N, D) bf16 via LDS-tiled transpose
// ---------------------------------------------------------------------------
__global__ void __launch_bounds__(256) prep_z(const float* __restrict__ z_e,
                                              unsigned short* __restrict__ z16) {
  __shared__ float tile[32][33];
  const int hw0 = blockIdx.x * 32;
  const int d0  = blockIdx.y * 32;
  const int b   = blockIdx.z;
  const int tx = threadIdx.x, ty = threadIdx.y;   // 32 x 8
  #pragma unroll
  for (int i = 0; i < 4; ++i) {
    const int d = d0 + ty + i * 8;
    tile[ty + i * 8][tx] = z_e[(b * DIM + d) * 1024 + hw0 + tx];
  }
  __syncthreads();
  #pragma unroll
  for (int i = 0; i < 4; ++i) {
    const int n = b * 1024 + hw0 + ty + i * 8;
    z16[n * DIM + d0 + tx] = f2bf(tile[tx][ty + i * 8]);
  }
}

// ---------------------------------------------------------------------------
// Stage DMA: 64 codes x 256 B (16 KB, contiguous in memory) -> padded LDS.
// TDM: one descriptor moves the whole tile, HW inserts 16 B pad per 256 B row
// (pad_interval=5 -> 64 dwords, pad_amount=3 -> 4 dwords => ROW_DW=68).
// Fallback: per-thread async b128 copies (ASYNCcnt), then plain loads.
// ---------------------------------------------------------------------------
__device__ __forceinline__ void stage_kick(const unsigned short* __restrict__ e16,
                                           unsigned int* __restrict__ buf,
                                           int code0, int tid) {
#if defined(USE_TDM)
  if (tid < 32) {   // wave 0 issues the tensor DMA for the whole block
    const unsigned long long ga =
        (unsigned long long)(const void*)(e16 + code0 * DIM);
    const unsigned int lds =
        (unsigned int)(unsigned long long)(lds_void*)buf;
    u32x4 g0;
    g0[0] = 1u;                                       // count=1, user descriptor
    g0[1] = lds;                                      // lds_addr (bytes)
    g0[2] = (unsigned int)ga;                         // global_addr[31:0]
    g0[3] = (unsigned int)((ga >> 32) & 0x01FFFFFFu)  // global_addr[56:32]
          | (2u << 30);                               // type = 2 ("image")
    i32x8 g1;
    g1[0] = (2 << 16)      // data_size = 4 B
          | (1 << 20)      // pad_enable
          | (5 << 22)      // pad_interval: 64 dwords
          | (3 << 25);     // pad_amount:   4 dwords
    g1[1] = (64 << 16);    // tensor_dim0 = 64 dwords/row
    g1[2] = (64 << 16);    // tensor_dim1 = 64 rows
    g1[3] = (64 << 16);    // tile_dim0   = 64 dwords
    g1[4] = 64;            // tile_dim1   = 64 rows (tile_dim2 = 0)
    g1[5] = 64;            // tensor_dim0_stride = 64 dwords (contiguous)
    g1[6] = 0;
    g1[7] = 0;
    const i32x4 z4 = {0, 0, 0, 0};
    const i32x8 z8 = {0, 0, 0, 0, 0, 0, 0, 0};
    __builtin_amdgcn_tensor_load_to_lds(g0, g1, z4, z4, z8, 0);
  }
#else
  #pragma unroll
  for (int i = 0; i < 4; ++i) {
    const int ch  = tid + i * 256;
    const int row = ch >> 4, col = ch & 15;
    const unsigned short* src = e16 + (code0 + row) * DIM + col * 8;
    unsigned int* dst = &buf[row * ROW_DW + col * 4];
#if defined(USE_ASYNC_LDS)
    __builtin_amdgcn_global_load_async_to_lds_b128(
        (global_v4i*)src, (lds_v4i*)dst, 0, 0);
#else
    *(uint4*)dst = *(const uint4*)src;
#endif
  }
#endif
}

__device__ __forceinline__ void stage_wait(int tid) {
#if defined(USE_TDM)
  if (tid < 32) __builtin_amdgcn_s_wait_tensorcnt(0);
#else
  WAIT_ASYNC();
#endif
}

// ---------------------------------------------------------------------------
// Kernel 3: fused distance-GEMM (bf16 WMMA, f32 acc) + argmin over K.
// Block = 256 threads = 8 waves; wave owns 16 rows; 256 blocks cover N.
// Codebook double-buffered through LDS, 64 codes (4 col-tiles) per stage.
// ---------------------------------------------------------------------------
__global__ void __launch_bounds__(256) vq_argmin(const unsigned short* __restrict__ z16,
                                                 const unsigned short* __restrict__ e16,
                                                 const float* __restrict__ enorm,
                                                 int* __restrict__ out_idx,
                                                 float* __restrict__ out_idx_f) {
  __shared__ unsigned int smem[2][STAGE_CODES * ROW_DW];   // 2 x 17,408 B
  const int tid  = threadIdx.x;
  const int lane = tid & 31;
  const int wave = tid >> 5;
  const int h = lane >> 4;      // half-wave selector (K/N split per ISA layout)
  const int m = lane & 15;      // A: row; B/C: column within tile
  const int row_base = blockIdx.x * 128 + wave * 16;

  // A fragments: 16 rows x 128 K, 4 fragments of K=32, ISA 16-bit A layout:
  // dwords 0..3 -> K = 8h + {0..7}; dwords 4..7 -> K = 16 + 8h + {0..7}
  FragA a[4];
  #pragma unroll
  for (int kc4 = 0; kc4 < 4; ++kc4) {
    const unsigned int* zr =
        (const unsigned int*)(z16 + (row_base + m) * DIM + kc4 * 32);
    #pragma unroll
    for (int v = 0; v < 4; ++v) a[kc4].u[v]     = zr[4 * h + v];
    #pragma unroll
    for (int v = 0; v < 4; ++v) a[kc4].u[4 + v] = zr[8 + 4 * h + v];
  }

  float best[8];
  int   bidx[8];
  #pragma unroll
  for (int r = 0; r < 8; ++r) { best[r] = 3.4e38f; bidx[r] = 0; }

  // Per-lane codebook norms for the current stage (columns t*16 + m).
  float curn[4];
  #pragma unroll
  for (int t = 0; t < 4; ++t) curn[t] = enorm[t * 16 + m];

  // Prologue: kick off stage 0.
  stage_kick(e16, smem[0], 0, tid);

  for (int stage = 0; stage < NSTAGES; ++stage) {
    stage_wait(tid);
    __syncthreads();   // stage data visible; previous compute finished everywhere

    float nextn[4] = {curn[0], curn[1], curn[2], curn[3]};
    if (stage + 1 < NSTAGES) {   // overlap next stage's DMA + norm loads
      stage_kick(e16, smem[(stage + 1) & 1], (stage + 1) * STAGE_CODES, tid);
      #pragma unroll
      for (int t = 0; t < 4; ++t)
        nextn[t] = enorm[(stage + 1) * STAGE_CODES + t * 16 + m];
    }

    const unsigned int* buf = smem[stage & 1];
    const int code0 = stage * STAGE_CODES;

    #pragma unroll
    for (int p = 0; p < 2; ++p) {     // two column tiles per step: dual chains
      const int t0 = 2 * p, t1 = t0 + 1;
      FragB b0[4], b1[4];
      #pragma unroll
      for (int kc4 = 0; kc4 < 4; ++kc4) {
        const uint4* r0 =
            (const uint4*)&buf[(t0 * 16 + m) * ROW_DW + kc4 * 16 + 4 * h];
        b0[kc4].q[0] = r0[0]; b0[kc4].q[1] = r0[2];
      }
      #pragma unroll
      for (int kc4 = 0; kc4 < 4; ++kc4) {
        const uint4* r1 =
            (const uint4*)&buf[(t1 * 16 + m) * ROW_DW + kc4 * 16 + 4 * h];
        b1[kc4].q[0] = r1[0]; b1[kc4].q[1] = r1[2];
      }
      f32x8 acc0 = {}, acc1 = {};
      #pragma unroll
      for (int kc4 = 0; kc4 < 4; ++kc4) {   // interleaved independent chains
        acc0 = __builtin_amdgcn_wmma_f32_16x16x32_bf16(
            false, a[kc4].v, false, b0[kc4].v, (short)0, acc0, false, false);
        acc1 = __builtin_amdgcn_wmma_f32_16x16x32_bf16(
            false, a[kc4].v, false, b1[kc4].v, (short)0, acc1, false, false);
      }
      // score = ||e||^2 - 2 z.e  (||z||^2 constant per row, drops from argmin)
      const int c0 = code0 + t0 * 16 + m, c1 = code0 + t1 * 16 + m;
      #pragma unroll
      for (int r = 0; r < 8; ++r) {
        const float s0 = fmaf(-2.0f, acc0[r], curn[t0]);
        if (s0 < best[r]) { best[r] = s0; bidx[r] = c0; }
        const float s1 = fmaf(-2.0f, acc1[r], curn[t1]);
        if (s1 < best[r]) { best[r] = s1; bidx[r] = c1; }
      }
    }

    #pragma unroll
    for (int t = 0; t < 4; ++t) curn[t] = nextn[t];
  }

  // Argmin across the 16 columns held per 16-lane half (C layout: lane = N).
  #pragma unroll
  for (int r = 0; r < 8; ++r) {
    float bv = best[r]; int bi = bidx[r];
    #pragma unroll
    for (int msk = 8; msk >= 1; msk >>= 1) {
      const float ov = __shfl_xor(bv, msk);
      const int   oi = __shfl_xor(bi, msk);
      if (ov < bv || (ov == bv && oi < bi)) { bv = ov; bi = oi; }
    }
    if (m == 0) {                       // lane 0 -> rows r, lane 16 -> rows 8+r
      const int row = row_base + h * 8 + r;
      out_idx[row]   = bi;
      out_idx_f[row] = (float)bi;
    }
  }
}

// ---------------------------------------------------------------------------
// Kernel 4: gather z_q into (B, D, H, W) + per-block MSE partial sums
// ---------------------------------------------------------------------------
__global__ void __launch_bounds__(256) gather_loss(const float* __restrict__ z_e,
                                                   const float* __restrict__ emb,
                                                   const int* __restrict__ idx,
                                                   float* __restrict__ out_q,
                                                   float* __restrict__ partial) {
  const int i  = blockIdx.x * 256 + threadIdx.x;
  const int hw = i & 1023;
  const int d  = (i >> 10) & 127;
  const int b  = i >> 17;
  const int n  = b * 1024 + hw;
  const int k  = idx[n];
  const float q = emb[k * DIM + d];
  out_q[i] = q;                                   // z_q_st forward value == z_q
  const float diff = q - z_e[i];
  float s = diff * diff;
  #pragma unroll
  for (int msk = 16; msk >= 1; msk >>= 1) s += __shfl_xor(s, msk);
  __shared__ float red[8];
  if ((threadIdx.x & 31) == 0) red[threadIdx.x >> 5] = s;
  __syncthreads();
  if (threadIdx.x == 0) {
    float t = 0.0f;
    #pragma unroll
    for (int w = 0; w < 8; ++w) t += red[w];
    partial[blockIdx.x] = t;
  }
}

// ---------------------------------------------------------------------------
// Kernel 5: reduce partials -> vq_loss = (1 + beta) * mean
// ---------------------------------------------------------------------------
__global__ void __launch_bounds__(256) final_loss(const float* __restrict__ partial,
                                                  float* __restrict__ out_loss) {
  float s = 0.0f;
  for (int i = threadIdx.x; i < OUTQ / 256; i += 256) s += partial[i];
  #pragma unroll
  for (int msk = 16; msk >= 1; msk >>= 1) s += __shfl_xor(s, msk);
  __shared__ float red[8];
  if ((threadIdx.x & 31) == 0) red[threadIdx.x >> 5] = s;
  __syncthreads();
  if (threadIdx.x == 0) {
    float t = 0.0f;
    #pragma unroll
    for (int w = 0; w < 8; ++w) t += red[w];
    out_loss[0] = 1.25f * t / (float)OUTQ;
  }
}

// ---------------------------------------------------------------------------
extern "C" void kernel_launch(void* const* d_in, const int* in_sizes, int n_in,
                              void* d_out, int out_size, void* d_ws, size_t ws_size,
                              hipStream_t stream) {
  (void)in_sizes; (void)n_in; (void)out_size; (void)ws_size;
  const float* z_e = (const float*)d_in[0];
  const float* emb = (const float*)d_in[1];
  float* out = (float*)d_out;

  char* w = (char*)d_ws;
  unsigned short* z16   = (unsigned short*)w;                                  // 8 MB
  unsigned short* e16   = (unsigned short*)(w + (8u << 20));                   // 512 KB
  float*          enorm = (float*)(w + (8u << 20) + (512u << 10));             // 8 KB
  int*            widx  = (int*)(w + (8u << 20) + (520u << 10));               // 128 KB
  float*          part  = (float*)(w + (8u << 20) + (648u << 10));             // 64 KB

  prep_emb<<<NUM_EMB, 128, 0, stream>>>(emb, e16, enorm);

  dim3 tgrid(32, 4, 32), tblk(32, 8, 1);
  prep_z<<<tgrid, tblk, 0, stream>>>(z_e, z16);

  vq_argmin<<<NVEC / 128, 256, 0, stream>>>(z16, e16, enorm, widx, out + IDXF_OFF);

  gather_loss<<<OUTQ / 256, 256, 0, stream>>>(z_e, emb, widx, out, part);

  final_loss<<<1, 256, 0, stream>>>(part, out + LOSS_OFF);
}
